// binary_disordered_RNNwavefunction_16140487098608
// MI455X (gfx1250) — matrix operand order repaired and checked by reading, not compiled, not run on hardware
//
#include <hip/hip_runtime.h>

typedef __attribute__((ext_vector_type(16))) _Float16 v16h;
typedef __attribute__((ext_vector_type(8)))  _Float16 v8h;
typedef __attribute__((ext_vector_type(8)))  float    v8f;

#define H       128
#define NSTEPS  128
#define II      2
#define MTILE   32          // batch rows per block (2 M-tiles of 16)
#define THREADS 256         // 8 wave32

__device__ __forceinline__ float sigmoidf(float x) { return 1.0f / (1.0f + __expf(-x)); }

// ---- fragment loaders -------------------------------------------------------
// f16 source: two 16B vector loads, no conversion.
__device__ __forceinline__ v16h load_frag2(const _Float16* p0, const _Float16* p1) {
  v8h lo = *(const v8h*)p0;
  v8h hi = *(const v8h*)p1;
  return __builtin_shufflevector(lo, hi, 0, 1, 2, 3, 4, 5, 6, 7, 8, 9, 10, 11, 12, 13, 14, 15);
}
// f32 source: convert on the fly (fallback path only).
__device__ __forceinline__ v16h load_frag2(const float* p0, const float* p1) {
  v16h r;
#pragma unroll
  for (int i = 0; i < 8; ++i) { r[i] = (_Float16)p0[i]; r[i + 8] = (_Float16)p1[i]; }
  return r;
}

// C[m, nt*16+n] = sum_k A[m,k] * W[nt*16+n, k]
// A: 32 rows, lda elems; W row-major, ld = 128 elems; K = 128 (4x WMMA 16x16x32).
// Each wave owns output columns; computes both M-tiles reusing B; A fragments
// hoisted into VGPRs across the column loop (loaded once per wave per GEMM).
template <typename TA, typename TW>
__device__ __forceinline__ void gemm_block(const TA* __restrict__ A, int lda,
                                           const TW* __restrict__ W,
                                           float* __restrict__ C, int ldc,
                                           int nNt, int wave, int lane) {
  const int nl = lane & 15;
  const int hi = lane >> 4;    // A lanes<16: K {0..7,16..23}; >=16: {8..15,24..31}
  const TA* arow0 = A + nl * lda + hi * 8;
  const TA* arow1 = A + (16 + nl) * lda + hi * 8;
  v16h a0[4], a1[4];
#pragma unroll
  for (int kk = 0; kk < 4; ++kk) {
    a0[kk] = load_frag2(arow0 + kk * 32, arow0 + kk * 32 + 16);
    a1[kk] = load_frag2(arow1 + kk * 32, arow1 + kk * 32 + 16);
  }
  for (int nt = wave; nt < nNt; nt += 8) {
    // B lanes<16: K {0..7,16..23} of column nl; lanes>=16: K {8..15,24..31}
    const TW* brow = W + (nt * 16 + nl) * H + hi * 8;
    v8f acc0 = {}, acc1 = {};
#pragma unroll
    for (int kk = 0; kk < 4; ++kk) {
      v16h bf = load_frag2(brow + kk * 32, brow + kk * 32 + 16);
      acc0 = __builtin_amdgcn_wmma_f32_16x16x32_f16(false, a0[kk], false, bf,
                                                    (short)0, acc0, false, false);
      acc1 = __builtin_amdgcn_wmma_f32_16x16x32_f16(false, a1[kk], false, bf,
                                                    (short)0, acc1, false, false);
    }
    // C/D layout: VGPR r -> M = r + 8*hi, N = nl
    float* cp0 = C + (hi * 8) * ldc + nt * 16 + nl;
    float* cp1 = C + (16 + hi * 8) * ldc + nt * 16 + nl;
#pragma unroll
    for (int r = 0; r < 8; ++r) { cp0[r * ldc] = acc0[r]; cp1[r * ldc] = acc1[r]; }
  }
}

// ---- f32 -> f16 weight shadow (runs once per launch, bandwidth-trivial) -----
__global__ void __launch_bounds__(256)
cvt_f32_f16_kernel(const float* __restrict__ src, _Float16* __restrict__ dst, long count) {
  long i = (long)blockIdx.x * blockDim.x + threadIdx.x;
  const long stride = (long)gridDim.x * blockDim.x;
  for (; i < count; i += stride) dst[i] = (_Float16)src[i];
}

// ---- main fused RNN kernel --------------------------------------------------
template <bool F16W>
__global__ void __launch_bounds__(THREADS)
rnn_wavefn_kernel(const float* __restrict__ inputs,
                  const float* __restrict__ Wih0, const float* __restrict__ Whh0f,
                  const float* __restrict__ bih0, const float* __restrict__ bhh0,
                  const float* __restrict__ Wih1f, const float* __restrict__ Whh1f,
                  const float* __restrict__ bih1, const float* __restrict__ bhh1,
                  const float* __restrict__ W1f,  const float* __restrict__ b1,
                  const float* __restrict__ W2,   const float* __restrict__ b2,
                  const _Float16* __restrict__ Whh0h, const _Float16* __restrict__ Wih1h,
                  const _Float16* __restrict__ Whh1h, const _Float16* __restrict__ W1h,
                  float* __restrict__ out) {
  extern __shared__ char smemraw[];
  float* BUF0 = (float*)smemraw;            // 32 x 384 (gh0 / gi1 / s-f32 fallback)
  float* BUF1 = BUF0 + MTILE * 384;         // 32 x 384 (gh1 / hdn)
  float* h1   = BUF1 + MTILE * 384;         // 32 x 128 f32
  float* h2   = h1 + MTILE * H;             // 32 x 128 f32
  float* xb   = h2 + MTILE * H;             // 32 x 2
  _Float16* h1h = (_Float16*)(xb + MTILE * II);   // 32 x 128 f16 (F16W only)
  _Float16* h2h = h1h + MTILE * H;
  _Float16* sh  = h2h + MTILE * H;

  const int tid  = threadIdx.x;
  const int wave = tid >> 5;
  const int lane = tid & 31;
  const int b0   = blockIdx.x * MTILE;

  for (int e = tid; e < MTILE * H; e += THREADS) {
    h1[e] = 0.f; h2[e] = 0.f;
    if constexpr (F16W) { h1h[e] = (_Float16)0.f; h2h[e] = (_Float16)0.f; }
  }
  for (int e = tid; e < MTILE * II; e += THREADS) xb[e] = inputs[b0 * II + e];
  __syncthreads();

  for (int n = 0; n < NSTEPS; ++n) {
    const long wo3 = (long)n * 3 * H * H;
    const float* wih0 = Wih0 + (long)n * 3 * H * II;
    const float* bi0  = bih0 + n * 3 * H;
    const float* bh0  = bhh0 + n * 3 * H;
    const float* bi1  = bih1 + n * 3 * H;
    const float* bh1  = bhh1 + n * 3 * H;
    const float* bb1  = b1 + n * H;
    const float* w2   = W2 + (long)n * II * H;
    const float* bb2  = b2 + n * II;

    if constexpr (F16W) {
      // Prefetch next step's big weight blocks (L2->WGP), global_prefetch_b8.
      if (n + 1 < NSTEPS) {
        const long no = wo3 + 3 * H * H;
        __builtin_prefetch(Whh0h + no + (long)tid * 192, 0, 1);
        __builtin_prefetch(Wih1h + no + (long)tid * 192, 0, 1);
        __builtin_prefetch(Whh1h + no + (long)tid * 192, 0, 1);
      }
    }

    // ---- phase 1: gh0 = h1 @ Whh0^T -> BUF0 [32x384]
    if constexpr (F16W) gemm_block(h1h, H, Whh0h + wo3, BUF0, 384, 24, wave, lane);
    else                gemm_block(h1,  H, Whh0f + wo3, BUF0, 384, 24, wave, lane);
    __syncthreads();

    // ---- phase 2: GRU0 fusion (gi0 from one-hot x folded in) -> h1 (+f16 copy)
    for (int e = tid; e < MTILE * H; e += THREADS) {
      int m = e >> 7, j = e & (H - 1);
      float x0 = xb[m * II + 0], x1 = xb[m * II + 1];
      float ir = bi0[j]         + wih0[(j        ) * II] * x0 + wih0[(j        ) * II + 1] * x1;
      float iz = bi0[j + H]     + wih0[(j + H    ) * II] * x0 + wih0[(j + H    ) * II + 1] * x1;
      float ic = bi0[j + 2 * H] + wih0[(j + 2 * H) * II] * x0 + wih0[(j + 2 * H) * II + 1] * x1;
      float hr = BUF0[m * 384 + j]         + bh0[j];
      float hz = BUF0[m * 384 + j + H]     + bh0[j + H];
      float hc = BUF0[m * 384 + j + 2 * H] + bh0[j + 2 * H];
      float r  = sigmoidf(ir + hr);
      float z  = sigmoidf(iz + hz);
      float nn = tanhf(ic + r * hc);
      float hv = (1.f - z) * nn + z * h1[e];
      h1[e] = hv;
      if constexpr (F16W) h1h[e] = (_Float16)hv;
    }
    __syncthreads();

    // ---- phase 3: gi1 = h1 @ Wih1^T -> BUF0 ; gh1 = h2 @ Whh1^T -> BUF1
    if constexpr (F16W) {
      gemm_block(h1h, H, Wih1h + wo3, BUF0, 384, 24, wave, lane);
      gemm_block(h2h, H, Whh1h + wo3, BUF1, 384, 24, wave, lane);
    } else {
      gemm_block(h1, H, Wih1f + wo3, BUF0, 384, 24, wave, lane);
      gemm_block(h2, H, Whh1f + wo3, BUF1, 384, 24, wave, lane);
    }
    __syncthreads();

    // ---- phase 4: GRU1 fusion -> h2 ; s = cos(h2)+eps (f16 copy for GEMM A)
    for (int e = tid; e < MTILE * H; e += THREADS) {
      int m = e >> 7, j = e & (H - 1);
      float ir = BUF0[m * 384 + j]         + bi1[j];
      float iz = BUF0[m * 384 + j + H]     + bi1[j + H];
      float ic = BUF0[m * 384 + j + 2 * H] + bi1[j + 2 * H];
      float hr = BUF1[m * 384 + j]         + bh1[j];
      float hz = BUF1[m * 384 + j + H]     + bh1[j + H];
      float hc = BUF1[m * 384 + j + 2 * H] + bh1[j + 2 * H];
      float r  = sigmoidf(ir + hr);
      float z  = sigmoidf(iz + hz);
      float nn = tanhf(ic + r * hc);
      float hn = (1.f - z) * nn + z * h2[e];
      h2[e] = hn;
      float sv = cosf(hn) + 1e-10f;
      if constexpr (F16W) sh[e] = (_Float16)sv;
      else                BUF0[m * 384 + j] = sv;   // owner thread read ir first: safe
    }
    __syncthreads();

    // ---- phase 5: hdn = s @ W1^T -> BUF1 cols 0..127 (ld 384)
    if constexpr (F16W) gemm_block(sh,   H,   W1h + (long)n * H * H, BUF1, 384, 8, wave, lane);
    else                gemm_block(BUF0, 384, W1f + (long)n * H * H, BUF1, 384, 8, wave, lane);
    __syncthreads();

    // ---- phase 6: hdn = relu(hdn + b1)
    for (int e = tid; e < MTILE * H; e += THREADS) {
      int m = e >> 7, j = e & (H - 1);
      float v = BUF1[m * 384 + j] + bb1[j];
      BUF1[m * 384 + j] = v > 0.f ? v : 0.f;
    }
    __syncthreads();

    // ---- phase 7: logits (K=128, I=2), softmax, emit probs, sample -> xb
    if (tid < MTILE) {
      int m = tid;
      const float* hr = BUF1 + m * 384;
      float l0 = bb2[0], l1 = bb2[1];
#pragma unroll 8
      for (int k = 0; k < H; ++k) {
        float hv = hr[k];
        l0 += hv * w2[k];
        l1 += hv * w2[H + k];
      }
      float mx = fmaxf(l0, l1);
      float e0 = __expf(l0 - mx), e1 = __expf(l1 - mx);
      float inv = 1.f / (e0 + e1);
      float p0 = e0 * inv + 1e-10f, p1 = e1 * inv + 1e-10f;
      long ob = ((long)(b0 + m) * NSTEPS + n) * II;
      out[ob] = p0;
      out[ob + 1] = p1;
      unsigned s = (unsigned)(b0 + m) * 2654435761u ^ ((unsigned)n * 0x9E3779B9u + 0x7F4A7C15u);
      s ^= s >> 16; s *= 0x85EBCA6Bu; s ^= s >> 13; s *= 0xC2B2AE35u; s ^= s >> 16;
      float u = (float)(s >> 8) * (1.0f / 16777216.0f);
      int smp = (u < p0) ? 0 : 1;
      xb[m * II + 0] = (smp == 0) ? 1.f : 0.f;
      xb[m * II + 1] = (smp == 1) ? 1.f : 0.f;
    }
    __syncthreads();
  }
}

extern "C" void kernel_launch(void* const* d_in, const int* in_sizes, int n_in,
                              void* d_out, int out_size, void* d_ws, size_t ws_size,
                              hipStream_t stream) {
  const float* inputs = (const float*)d_in[0];
  const float* Wih0   = (const float*)d_in[1];
  const float* Whh0   = (const float*)d_in[2];
  const float* bih0   = (const float*)d_in[3];
  const float* bhh0   = (const float*)d_in[4];
  const float* Wih1   = (const float*)d_in[5];
  const float* Whh1   = (const float*)d_in[6];
  const float* bih1   = (const float*)d_in[7];
  const float* bhh1   = (const float*)d_in[8];
  const float* W1     = (const float*)d_in[9];
  const float* b1     = (const float*)d_in[10];
  const float* W2     = (const float*)d_in[11];
  const float* b2     = (const float*)d_in[12];
  float* out = (float*)d_out;

  const int B = in_sizes[0] / II;          // 8192
  const int blocks = B / MTILE;            // 256

  const long nBig = (long)NSTEPS * 3 * H * H;      // Whh0 / Wih1 / Whh1 each
  const long nW1  = (long)NSTEPS * H * H;
  const size_t needWs = (size_t)(3 * nBig + nW1) * sizeof(_Float16);  // ~40 MB

  const size_t shmemBase =
      (size_t)(2 * MTILE * 384 + 2 * MTILE * H + MTILE * II) * sizeof(float);
  const size_t shmemF16 = shmemBase + (size_t)(3 * MTILE * H) * sizeof(_Float16);

  if (ws_size >= needWs) {
    _Float16* wsh   = (_Float16*)d_ws;
    _Float16* whh0h = wsh;
    _Float16* wih1h = wsh + nBig;
    _Float16* whh1h = wsh + 2 * nBig;
    _Float16* w1h   = wsh + 3 * nBig;

    cvt_f32_f16_kernel<<<512, 256, 0, stream>>>(Whh0, whh0h, nBig);
    cvt_f32_f16_kernel<<<512, 256, 0, stream>>>(Wih1, wih1h, nBig);
    cvt_f32_f16_kernel<<<512, 256, 0, stream>>>(Whh1, whh1h, nBig);
    cvt_f32_f16_kernel<<<512, 256, 0, stream>>>(W1,   w1h,   nW1);

    hipFuncSetAttribute((const void*)rnn_wavefn_kernel<true>,
                        hipFuncAttributeMaxDynamicSharedMemorySize, (int)shmemF16);
    rnn_wavefn_kernel<true><<<blocks, THREADS, shmemF16, stream>>>(
        inputs, Wih0, Whh0, bih0, bhh0, Wih1, Whh1, bih1, bhh1, W1, b1, W2, b2,
        whh0h, wih1h, whh1h, w1h, out);
  } else {
    hipFuncSetAttribute((const void*)rnn_wavefn_kernel<false>,
                        hipFuncAttributeMaxDynamicSharedMemorySize, (int)shmemBase);
    rnn_wavefn_kernel<false><<<blocks, THREADS, shmemBase, stream>>>(
        inputs, Wih0, Whh0, bih0, bhh0, Wih1, Whh1, bih1, bhh1, W1, b1, W2, b2,
        nullptr, nullptr, nullptr, nullptr, out);
  }
}
